// FacePartGAT_67370857005430
// MI455X (gfx1250) — compile-verified, matrix operand order checked by reading, not run on hardware
//
#include <hip/hip_runtime.h>

// FacePartGAT on MI455X (gfx1250, wave32, WMMA).
// Compute-bound (~25 GFLOP, ~18MB working set, L2-resident): all matmuls via
// v_wmma_f32_16x16x32_bf16. Layouts: A row-major, B pre-transposed, C stored
// transposed -> every fragment transfer is contiguous 128-bit.
// Attention: out^T = H^T @ alpha^T, alpha B-fragment synthesized in registers.
// The H^T panel is DMA'd chunk-wise into double-buffered LDS with
// GLOBAL_LOAD_ASYNC_TO_LDS_B128 (ASYNCcnt) so the next chunk's copy overlaps
// this chunk's WMMA work; 8 waves/WG share each chunk (8x traffic cut).
// Softmax denominator folds into the same pass (per-lane bf16 mass + shfl_xor).

#define NN   4096   // nodes
#define HIDD 768    // hidden dim
#define GHC  128    // gat hidden per head
#define HEADS1 4
#define HG   (HEADS1*GHC)   // 512

#define IC     128          // staged source-chunk width (nodes)
#define LDSROW (IC + 8)     // +16B pad: row stride 272B == 16 mod 32 -> conflict-free b128

typedef __attribute__((ext_vector_type(16))) __bf16 v16bf;
typedef __attribute__((ext_vector_type(8)))  __bf16 v8bf;
typedef __attribute__((ext_vector_type(8)))  float  v8f;

static __device__ __forceinline__ float lrelu(float x) { return x > 0.f ? x : 0.2f * x; }
static __device__ __forceinline__ float elu1(float x)  { return x > 0.f ? x : (__expf(x) - 1.f); }

// ---- WMMA fragment helper (ISA 7.12.2, wave32) -----------------------------
// 16-bit A 16x32 (and mirrored B 32x16): lanes 0-15 own row/col (lane&15) with
// K runs {0..7,16..23}; lanes 16-31 same row/col, K {8..15,24..31}.
static __device__ __forceinline__
v16bf load_frag(const __bf16* __restrict__ P, int ld, int r0, int k0, int lane) {
    const __bf16* p = P + (size_t)(r0 + (lane & 15)) * ld + k0 + ((lane & 16) ? 8 : 0);
    v8bf lo = *(const v8bf*)p;
    v8bf hi = *(const v8bf*)(p + 16);
    return __builtin_shufflevector(lo, hi, 0, 1, 2, 3, 4, 5, 6, 7,
                                   8, 9, 10, 11, 12, 13, 14, 15);
}

// ---------------- kernels ---------------------------------------------------

__global__ void cvt_f32_bf16(const float* __restrict__ src, __bf16* __restrict__ dst, int n) {
    int i = blockIdx.x * blockDim.x + threadIdx.x;
    if (i < n) dst[i] = (__bf16)src[i];
}

// src [R][C] f32 -> dst [C][R] bf16 (one-time weight packing)
__global__ void cvtT_f32_bf16(const float* __restrict__ src, __bf16* __restrict__ dst,
                              int R, int C) {
    int idx = blockIdx.x * blockDim.x + threadIdx.x;
    if (idx >= R * C) return;
    int r = idx / C, c = idx % C;
    dst[(size_t)c * R + r] = (__bf16)src[idx];
}

// Ct[N][M] = (A[M,K] @ Bt[N,K]^T)^T. One wave: 32 rows x 64 cols
// (2 A-frags x 4 B-frags -> 8 WMMA per 12 b128 loads).
__global__ void gemm_bf16_tn(const __bf16* __restrict__ A, const __bf16* __restrict__ Bt,
                             __bf16* __restrict__ Ct, int M, int N, int K) {
    constexpr int MT = 2, NT = 4;
    int gwave = (blockIdx.x * blockDim.x + threadIdx.x) >> 5;
    int lane  = threadIdx.x & 31;
    int nt64  = N >> 6;
    int tm = gwave / nt64, tn = gwave % nt64;
    if (tm >= (M >> 5)) return;
    int m0 = tm * 32, n0 = tn * 64;

    v8f acc[MT][NT];
#pragma unroll
    for (int s = 0; s < MT; ++s)
#pragma unroll
        for (int t = 0; t < NT; ++t)
            acc[s][t] = (v8f){0.f, 0.f, 0.f, 0.f, 0.f, 0.f, 0.f, 0.f};

    for (int k = 0; k < K; k += 32) {
        v16bf a0 = load_frag(A, K, m0,      k, lane);
        v16bf a1 = load_frag(A, K, m0 + 16, k, lane);
#pragma unroll
        for (int t = 0; t < NT; ++t) {
            v16bf b = load_frag(Bt, K, n0 + t * 16, k, lane);
            acc[0][t] = __builtin_amdgcn_wmma_f32_16x16x32_bf16(false, a0, false, b,
                                                                (short)0, acc[0][t], false, false);
            acc[1][t] = __builtin_amdgcn_wmma_f32_16x16x32_bf16(false, a1, false, b,
                                                                (short)0, acc[1][t], false, false);
        }
    }
#pragma unroll
    for (int s = 0; s < MT; ++s) {
        int mbase = m0 + s * 16 + ((lane & 16) ? 8 : 0);
#pragma unroll
        for (int t = 0; t < NT; ++t) {
            int col = n0 + t * 16 + (lane & 15);
            v8bf p;
#pragma unroll
            for (int r = 0; r < 8; ++r) p[r] = (__bf16)acc[s][t][r];
            *(v8bf*)(Ct + (size_t)col * M + mbase) = p;   // contiguous 16B store
        }
    }
}

// es[h][n] = <HT[h*ch+c][n], a_src[h][c]>_c ; ed likewise (coalesced over nodes).
__global__ void scores_k(const __bf16* __restrict__ HT, const float* __restrict__ a_src,
                         const float* __restrict__ a_dst, float* __restrict__ es,
                         float* __restrict__ ed, int n, int heads, int ch) {
    int node = blockIdx.x * blockDim.x + threadIdx.x;
    if (node >= n) return;
    for (int h = 0; h < heads; ++h) {
        float s = 0.f, d = 0.f;
        for (int c = 0; c < ch; ++c) {
            float v = (float)HT[(size_t)(h * ch + c) * n + node];
            s += v * a_src[h * ch + c];
            d += v * a_dst[h * ch + c];
        }
        es[(size_t)h * n + node] = s;
        ed[(size_t)h * n + node] = d;
    }
}

// Mh[h] = max_n es[h][n]. lrelu monotone => exact softmax max = lrelu(ed+Mh).
__global__ void colmax_k(const float* __restrict__ es, float* __restrict__ Mh, int n) {
    __shared__ float red[256];
    int h = blockIdx.x;
    float m = -3.0e38f;
    for (int i = threadIdx.x; i < n; i += blockDim.x)
        m = fmaxf(m, es[(size_t)h * n + i]);
    red[threadIdx.x] = m;
    __syncthreads();
    for (int s = 128; s > 0; s >>= 1) {
        if ((int)threadIdx.x < s) red[threadIdx.x] = fmaxf(red[threadIdx.x], red[threadIdx.x + s]);
        __syncthreads();
    }
    if (threadIdx.x == 0) Mh[h] = red[0];
}

// Async DMA of one H^T chunk into an LDS buffer: 16B per lane per issue,
// tracked by ASYNCcnt (no VGPR data path). LDS byte address = low 32 bits of
// the generic pointer (aperture keeps the wave-relative offset there).
static __device__ __forceinline__
void async_copy_chunk(const __bf16* __restrict__ HT, __bf16* buf,
                      int hbase, int n, int i0, int tid) {
#pragma unroll
    for (int w = 0; w < (GHC * IC / 8) / 256; ++w) {   // 8 issues/thread
        int id  = tid + 256 * w;
        int row = id >> 4;                             // IC/8 = 16 segs/row
        int seg = id & 15;
        unsigned lds = (unsigned)(size_t)(void*)(buf + row * LDSROW + seg * 8);
        const __bf16* g = HT + (size_t)(hbase + row) * n + i0 + seg * 8;
        asm volatile("global_load_async_to_lds_b128 %0, %1, off"
                     :: "v"(lds), "v"(g) : "memory");
    }
}

// Fused attention + softmax denominator, transposed orientation:
// out^T[c][j] = (1/Z_j) sum_i HT[c][i] * w[i][j],  w = exp(lrelu(ed_j+es_i)-mstar_j).
// Double-buffered async-LDS pipeline: wait(chunk c) -> barrier -> issue DMA of
// chunk c+1 -> WMMA on chunk c. One barrier per chunk.
__global__ void __launch_bounds__(256)
gat_attn_k(const __bf16* __restrict__ HT, const float* __restrict__ es,
           const float* __restrict__ ed, const float* __restrict__ Mh,
           const float* __restrict__ bias,
           __bf16* __restrict__ Xout,   // [n][heads*GHC] bf16 or null
           float*  __restrict__ XoutF,  // [n][GHC] f32 (heads==1) or null
           int n, int heads) {
    constexpr int CF = GHC / 16;  // 8 channel tiles
    __shared__ __bf16 tile[2][GHC * LDSROW];   // 2 x 34.8 KB

    int tid   = threadIdx.x;
    int lane  = tid & 31;
    int wv    = tid >> 5;                 // wave in block: 0..7
    int jb    = blockIdx.x / heads;
    int h     = blockIdx.x % heads;
    int j0    = jb * 128 + wv * 16;
    int hbase = h * GHC;

    int   jlane = j0 + (lane & 15);
    float edj   = ed[(size_t)h * n + jlane];
    float mstar = lrelu(edj + Mh[h]);
    int   kb    = (lane & 16) ? 8 : 0;
    const float* esh = es + (size_t)h * n;

    v8f acc[CF];
#pragma unroll
    for (int f = 0; f < CF; ++f) acc[f] = (v8f){0.f, 0.f, 0.f, 0.f, 0.f, 0.f, 0.f, 0.f};
    float zpart = 0.f;

    int nch = n / IC;
    async_copy_chunk(HT, tile[0], hbase, n, 0, tid);

    for (int c = 0; c < nch; ++c) {
        asm volatile("s_wait_asynccnt 0x0" ::: "memory");  // my DMA for chunk c done
        __syncthreads();                                   // everyone's DMA done; prev reads done
        if (c + 1 < nch)
            async_copy_chunk(HT, tile[(c + 1) & 1], hbase, n, (c + 1) * IC, tid);

        const __bf16* buf = tile[c & 1];
        int i0 = c * IC;
#pragma unroll
        for (int ii = 0; ii < IC; ii += 32) {
            v16bf b;
#pragma unroll
            for (int u = 0; u < 8; ++u) {
                __bf16 b0 = (__bf16)__expf(lrelu(edj + esh[i0 + ii + kb + u])      - mstar);
                __bf16 b1 = (__bf16)__expf(lrelu(edj + esh[i0 + ii + 16 + kb + u]) - mstar);
                b[u] = b0; b[u + 8] = b1;
                zpart += (float)b0 + (float)b1;   // exact bf16 mass fed to WMMA
            }
#pragma unroll
            for (int f = 0; f < CF; ++f) {
                v16bf a = load_frag(buf, LDSROW, f * 16, ii, lane);  // ds_load_b128 x2
                acc[f] = __builtin_amdgcn_wmma_f32_16x16x32_bf16(false, a, false, b,
                                                                 (short)0, acc[f], false, false);
            }
        }
    }

    // Z_j: this lane + partner lane (lane^16) together saw every source once.
    float z    = zpart + __shfl_xor(zpart, 16, 32);
    float zinv = 1.f / z;

    int chalf = (lane & 16) ? 8 : 0;
#pragma unroll
    for (int f = 0; f < CF; ++f) {
        v8bf packed;
#pragma unroll
        for (int r = 0; r < 8; ++r) {
            int c = f * 16 + chalf + r;                 // channel within head
            float v = acc[f][r] * zinv + bias[h * GHC + c];
            v = elu1(v);
            packed[r] = (__bf16)v;
            if (XoutF) XoutF[(size_t)jlane * GHC + c] = v;
        }
        if (Xout)
            *(v8bf*)(Xout + (size_t)jlane * (heads * GHC) + h * GHC + f * 16 + chalf) = packed;
    }
}

// mean[c] = (1/n) * sum_j X3[j][c]; one block per channel.
__global__ void mean_k(const float* __restrict__ X, float* __restrict__ mean, int n, int ch) {
    __shared__ float red[256];
    int c = blockIdx.x;
    float s = 0.f;
    for (int j = threadIdx.x; j < n; j += blockDim.x) s += X[(size_t)j * ch + c];
    red[threadIdx.x] = s;
    __syncthreads();
    for (int t = 128; t > 0; t >>= 1) {
        if ((int)threadIdx.x < t) red[threadIdx.x] += red[threadIdx.x + t];
        __syncthreads();
    }
    if (threadIdx.x == 0) mean[c] = red[0] / (float)n;
}

// out[o] = <mean, Wf[:,o]> + bf[o]
__global__ void final_fc_k(const float* __restrict__ mean, const float* __restrict__ Wf,
                           const float* __restrict__ bfv, float* __restrict__ out,
                           int ch, int hid) {
    int o = blockIdx.x * blockDim.x + threadIdx.x;
    if (o >= hid) return;
    float s = bfv[o];
    for (int c = 0; c < ch; ++c) s += mean[c] * Wf[(size_t)c * hid + o];
    out[o] = s;
}

// ---------------- host launch ----------------------------------------------

extern "C" void kernel_launch(void* const* d_in, const int* in_sizes, int n_in,
                              void* d_out, int out_size, void* d_ws, size_t ws_size,
                              hipStream_t stream) {
    (void)in_sizes; (void)n_in; (void)out_size; (void)ws_size;
    const float* roi = (const float*)d_in[0];
    const float* W1  = (const float*)d_in[1];
    const float* as1 = (const float*)d_in[2];
    const float* ad1 = (const float*)d_in[3];
    const float* b1  = (const float*)d_in[4];
    const float* W2  = (const float*)d_in[5];
    const float* as2 = (const float*)d_in[6];
    const float* ad2 = (const float*)d_in[7];
    const float* b2  = (const float*)d_in[8];
    const float* Wf  = (const float*)d_in[9];
    const float* bfv = (const float*)d_in[10];
    float* out = (float*)d_out;

    char* w = (char*)d_ws;
    size_t off = 0;
    auto alloc = [&](size_t bytes) -> char* {
        char* p = w + off;
        off = (off + bytes + 255) & ~(size_t)255;
        return p;
    };

    __bf16* Xbf  = (__bf16*)alloc((size_t)NN * HIDD * 2);  // [NN][HIDD]
    __bf16* W1T  = (__bf16*)alloc((size_t)HG * HIDD * 2);  // [HG][HIDD]
    __bf16* W2T  = (__bf16*)alloc((size_t)GHC * HG * 2);   // [GHC][HG]
    __bf16* H1T  = (__bf16*)alloc((size_t)HG * NN * 2);    // [HG][NN]
    __bf16* X2b  = (__bf16*)alloc((size_t)NN * HG * 2);    // [NN][HG]
    __bf16* H2T  = (__bf16*)alloc((size_t)GHC * NN * 2);   // [GHC][NN]
    float*  X3f  = (float*)alloc((size_t)NN * GHC * 4);    // [NN][GHC]
    float*  es1  = (float*)alloc((size_t)HEADS1 * NN * 4);
    float*  ed1  = (float*)alloc((size_t)HEADS1 * NN * 4);
    float*  M1   = (float*)alloc(HEADS1 * 4);
    float*  es2  = (float*)alloc((size_t)NN * 4);
    float*  ed2  = (float*)alloc((size_t)NN * 4);
    float*  M2   = (float*)alloc(4);
    float*  mean = (float*)alloc(GHC * 4);

    const int TB = 256;  // 8 waves / block
    auto cdiv = [](int a, int b) { return (a + b - 1) / b; };

    // pack inputs to bf16 (weights transposed for B-operand layout)
    cvt_f32_bf16<<<cdiv(NN * HIDD, TB), TB, 0, stream>>>(roi, Xbf, NN * HIDD);
    cvtT_f32_bf16<<<cdiv(HIDD * HG, TB), TB, 0, stream>>>(W1, W1T, HIDD, HG);
    cvtT_f32_bf16<<<cdiv(HG * GHC, TB), TB, 0, stream>>>(W2, W2T, HG, GHC);

    // ---- Layer 1 ----
    gemm_bf16_tn<<<cdiv((NN / 32) * (HG / 64), 8), TB, 0, stream>>>(Xbf, W1T, H1T, NN, HG, HIDD);
    scores_k<<<cdiv(NN, TB), TB, 0, stream>>>(H1T, as1, ad1, es1, ed1, NN, HEADS1, GHC);
    colmax_k<<<HEADS1, TB, 0, stream>>>(es1, M1, NN);
    gat_attn_k<<<(NN / 128) * HEADS1, TB, 0, stream>>>(H1T, es1, ed1, M1, b1,
                                                       X2b, nullptr, NN, HEADS1);

    // ---- Layer 2 (1 head) ----
    gemm_bf16_tn<<<cdiv((NN / 32) * (GHC / 64), 8), TB, 0, stream>>>(X2b, W2T, H2T, NN, GHC, HG);
    scores_k<<<cdiv(NN, TB), TB, 0, stream>>>(H2T, as2, ad2, es2, ed2, NN, 1, GHC);
    colmax_k<<<1, TB, 0, stream>>>(es2, M2, NN);
    gat_attn_k<<<NN / 128, TB, 0, stream>>>(H2T, es2, ed2, M2, b2,
                                            nullptr, X3f, NN, 1);

    // ---- Mean + Final FC ----
    mean_k<<<GHC, TB, 0, stream>>>(X3f, mean, NN, GHC);
    final_fc_k<<<cdiv(HIDD, TB), TB, 0, stream>>>(mean, Wf, bfv, out, GHC, HIDD);
}
